// ParticleFlowNetwork_33088428048678
// MI455X (gfx1250) — compile-verified, matrix operand order checked by reading, not run on hardware
//
#include <hip/hip_runtime.h>
#include <hip/hip_bf16.h>
#include <math.h>

typedef __attribute__((ext_vector_type(16))) _Float16 v16h;
typedef __attribute__((ext_vector_type(8)))  float    v8f;

#define H_PAD_N 112   // 100 -> 7 tiles of 16 (N dimension)
#define H_PAD_K 128   // 100 -> 4 tiles of 32 (K dimension)
#define ACT_STRIDE 128

// ---------------------------------------------------------------------------
// Fragment loader. CDNA5 16-bit A-matrix 16x32 layout (ISA 7.12.2):
//   lanes 0-15 : M = lane, VGPR v<4 -> K = 2v,2v+1 ; v>=4 -> K = 16+2(v-4)..
//   lanes 16-31: same M, K base shifted by +8.
// Weights are stored transposed (Wt[n][k], stride = K-pad) so the B fragment
// uses the identical gather with n in place of m; each half-fragment is a
// contiguous 16-byte run -> 2x ds_load_b128 per fragment (verified in asm).
// ---------------------------------------------------------------------------
__device__ __forceinline__ v16h load_frag(const _Float16* __restrict__ p,
                                          int stride, int row, int kk, bool hi) {
    v16h f;
    const int kb = hi ? 8 : 0;
    const _Float16* base = p + row * stride + kk * 32;
#pragma unroll
    for (int v = 0; v < 8; ++v) {
        int k = (v < 4) ? (kb + 2 * v) : (16 + kb + 2 * (v - 4));
        f[2 * v]     = base[k];
        f[2 * v + 1] = base[k + 1];
    }
    return f;
}

// One dense layer on a wave-private 16x128 f16 activation tile in LDS.
// All NT output-tile accumulators stay in registers across the K loop, so the
// (f16) outputs can be written back into the same LDS tile afterwards
// (same-wave LDS ops are kept in order; compiler inserts s_wait_dscnt).
template <int NT, int KT, bool RELU>
__device__ __forceinline__ void dense_layer(_Float16* __restrict__ act,
                                            const _Float16* __restrict__ Wt,
                                            const float* __restrict__ bias,
                                            int nreal, int lane) {
    const bool hi = lane >= 16;
    const int  lr = lane & 15;
    v8f acc[NT] = {};
#pragma unroll
    for (int kk = 0; kk < KT; ++kk) {
        v16h a = load_frag(act, ACT_STRIDE, lr, kk, hi);
#pragma unroll
        for (int j = 0; j < NT; ++j) {
            v16h b = load_frag(Wt, KT * 32, j * 16 + lr, kk, hi);
            acc[j] = __builtin_amdgcn_wmma_f32_16x16x32_f16(
                false, a, false, b, (short)0, acc[j], false, false);
        }
    }
#pragma unroll
    for (int j = 0; j < NT; ++j) {
        int   n  = j * 16 + lr;
        float bv = (n < nreal) ? bias[n] : 0.0f;
#pragma unroll
        for (int v = 0; v < 8; ++v) {
            float o = acc[j][v] + bv;
            if (RELU) o = fmaxf(o, 0.0f);
            int m = v + (hi ? 8 : 0);
            act[m * ACT_STRIDE + n] = (_Float16)o;
        }
    }
}

// ---------------------------------------------------------------------------
// Stage 1: per-particle MLP; each 16-row tile's column sums (incl. bias) go to
// pbuf[tile][8] (deterministic, no atomics). Persistent blocks: 512 blocks x
// 8 waves, each wave processes 8 tiles -> weight conversion amortized 8x.
// ---------------------------------------------------------------------------
__global__ void pfn_stage1(const float* __restrict__ x,
                           const float* __restrict__ W0, const float* __restrict__ b0,
                           const float* __restrict__ W1, const float* __restrict__ b1,
                           const float* __restrict__ W2, const float* __restrict__ b2,
                           const float* __restrict__ W3, const float* __restrict__ b3,
                           float* __restrict__ pbuf) {
    extern __shared__ _Float16 smem[];
    _Float16* sW0t = smem;                       // [112][32]
    _Float16* sW1t = sW0t + H_PAD_N * 32;        // [112][128]
    _Float16* sW2t = sW1t + H_PAD_N * H_PAD_K;   // [112][128]
    _Float16* sW3t = sW2t + H_PAD_N * H_PAD_K;   // [16][128]
    _Float16* sAct = sW3t + 16 * H_PAD_K;        // 8 waves x 16 x 128

    const int tid = threadIdx.x;
    // Convert weights to f16, transposed + zero-padded.
    for (int i = tid; i < H_PAD_N * 32; i += 256) {
        int n = i / 32, k = i % 32;
        sW0t[i] = (k < 4 && n < 100) ? (_Float16)W0[k * 100 + n] : (_Float16)0.0f;
    }
    for (int i = tid; i < H_PAD_N * H_PAD_K; i += 256) {
        int n = i / H_PAD_K, k = i % H_PAD_K;
        _Float16 z = (_Float16)0.0f;
        sW1t[i] = (k < 100 && n < 100) ? (_Float16)W1[k * 100 + n] : z;
        sW2t[i] = (k < 100 && n < 100) ? (_Float16)W2[k * 100 + n] : z;
    }
    for (int i = tid; i < 16 * H_PAD_K; i += 256) {
        int n = i / H_PAD_K, k = i % H_PAD_K;
        sW3t[i] = (k < 100 && n < 8) ? (_Float16)W3[k * 8 + n] : (_Float16)0.0f;
    }
    for (int i = tid; i < 8 * 16 * ACT_STRIDE; i += 256) sAct[i] = (_Float16)0.0f;
    __syncthreads();

    const int wave = tid >> 5;
    const int lane = tid & 31;
    const bool hi  = lane >= 16;
    const int  lr  = lane & 15;
    _Float16* act  = sAct + wave * 16 * ACT_STRIDE;

    const int base = blockIdx.x * 8 + wave;      // 4096 persistent waves

    for (int it = 0; it < 8; ++it) {
        const long tile = (long)base + (long)it * 4096;  // 32768 tiles total
        const long r0   = tile * 16;                     // first row of B*P

        // Stage x (K=4) into cols 0..3 and re-zero cols 4..31 (layer 0 only
        // reads K in [0,32); cols 32..127 are overwritten/zero-invariant).
        if (lane < 16) {
            const float* xr = x + (size_t)(r0 + lane) * 4;
            _Float16* ar = act + lane * ACT_STRIDE;
#pragma unroll
            for (int k = 0; k < 4; ++k) ar[k] = (_Float16)xr[k];
#pragma unroll
            for (int k = 4; k < 32; ++k) ar[k] = (_Float16)0.0f;
        }

        dense_layer<7, 1, false>(act, sW0t, b0, 100, lane);  // Linear(4,100)
        dense_layer<7, 4, true >(act, sW1t, b1, 100, lane);  // +ReLU
        dense_layer<7, 4, true >(act, sW2t, b2, 100, lane);  // +ReLU

        // Final Linear(100,8): per-row bias, sum the 16 rows, store partial.
        v8f acc = {};
#pragma unroll
        for (int kk = 0; kk < 4; ++kk) {
            v16h a = load_frag(act, ACT_STRIDE, lr, kk, hi);
            v16h b = load_frag(sW3t, H_PAD_K, lr, kk, hi);
            acc = __builtin_amdgcn_wmma_f32_16x16x32_f16(
                false, a, false, b, (short)0, acc, false, false);
        }
        float bv = (lr < 8) ? b3[lr] : 0.0f;
        float s  = 8.0f * bv;                // bias for this lane's 8 rows
#pragma unroll
        for (int v = 0; v < 8; ++v) s += acc[v];
        s += __shfl_xor(s, 16);              // rows 0-7 + rows 8-15
        if (lane < 8) pbuf[tile * 8 + lane] = s;
    }
}

// ---------------------------------------------------------------------------
// Stage 2: reduce the 8 tile-partials per event into lat[16][8] tiles, then
// run the event head and the 2-way softmax. 32 blocks x 8 waves = 256 tiles.
// ---------------------------------------------------------------------------
__global__ void pfn_stage2(const float* __restrict__ pbuf,
                           const float* __restrict__ U0, const float* __restrict__ c0,
                           const float* __restrict__ U1, const float* __restrict__ c1,
                           const float* __restrict__ U2, const float* __restrict__ c2,
                           const float* __restrict__ U3, const float* __restrict__ c3,
                           const float* __restrict__ U4, const float* __restrict__ c4,
                           float* __restrict__ out) {
    extern __shared__ _Float16 smem[];
    _Float16* sU0t = smem;                       // [112][32]
    _Float16* sU1t = sU0t + H_PAD_N * 32;        // [112][128]
    _Float16* sU2t = sU1t + H_PAD_N * H_PAD_K;
    _Float16* sU3t = sU2t + H_PAD_N * H_PAD_K;
    _Float16* sU4t = sU3t + H_PAD_N * H_PAD_K;   // [16][128]
    _Float16* sAct = sU4t + 16 * H_PAD_K;

    const int tid = threadIdx.x;
    for (int i = tid; i < H_PAD_N * 32; i += 256) {
        int n = i / 32, k = i % 32;
        sU0t[i] = (k < 8 && n < 100) ? (_Float16)U0[k * 100 + n] : (_Float16)0.0f;
    }
    for (int i = tid; i < H_PAD_N * H_PAD_K; i += 256) {
        int n = i / H_PAD_K, k = i % H_PAD_K;
        _Float16 z = (_Float16)0.0f;
        bool in = (k < 100 && n < 100);
        sU1t[i] = in ? (_Float16)U1[k * 100 + n] : z;
        sU2t[i] = in ? (_Float16)U2[k * 100 + n] : z;
        sU3t[i] = in ? (_Float16)U3[k * 100 + n] : z;
    }
    for (int i = tid; i < 16 * H_PAD_K; i += 256) {
        int n = i / H_PAD_K, k = i % H_PAD_K;
        sU4t[i] = (k < 100 && n < 2) ? (_Float16)U4[k * 2 + n] : (_Float16)0.0f;
    }
    for (int i = tid; i < 8 * 16 * ACT_STRIDE; i += 256) sAct[i] = (_Float16)0.0f;
    __syncthreads();

    const int wave = tid >> 5;
    const int lane = tid & 31;
    const bool hi  = lane >= 16;
    const int  lr  = lane & 15;
    _Float16* act  = sAct + wave * 16 * ACT_STRIDE;

    const int tile = blockIdx.x * 8 + wave;      // 256 tiles over 4096 events
    const int r0   = tile * 16;

    // Deterministic sum-pool: event b's latent = sum of its 8 tile partials.
    if (lane < 16) {
        const float* pr = pbuf + (size_t)(r0 + lane) * 64;  // 8 tiles x 8 lat
        _Float16* ar = act + lane * ACT_STRIDE;
#pragma unroll
        for (int k = 0; k < 8; ++k) {
            float s = 0.0f;
#pragma unroll
            for (int p = 0; p < 8; ++p) s += pr[p * 8 + k];
            ar[k] = (_Float16)s;
        }
    }

    dense_layer<7, 1, true>(act, sU0t, c0, 100, lane);
    dense_layer<7, 4, true>(act, sU1t, c1, 100, lane);
    dense_layer<7, 4, true>(act, sU2t, c2, 100, lane);
    dense_layer<7, 4, true>(act, sU3t, c3, 100, lane);

    // Final Linear(100,2) + softmax over the 2 logits of each row.
    {
        v8f acc = {};
#pragma unroll
        for (int kk = 0; kk < 4; ++kk) {
            v16h a = load_frag(act, ACT_STRIDE, lr, kk, hi);
            v16h b = load_frag(sU4t, H_PAD_K, lr, kk, hi);
            acc = __builtin_amdgcn_wmma_f32_16x16x32_f16(
                false, a, false, b, (short)0, acc, false, false);
        }
        float bv = (lr < 2) ? c4[lr] : 0.0f;
#pragma unroll
        for (int v = 0; v < 8; ++v) {
            float logit = acc[v] + bv;
            float other = __shfl_xor(logit, 1);   // lanes (n=0)<->(n=1) pair
            if (lr < 2) {
                int   m   = v + (hi ? 8 : 0);
                int   row = r0 + m;
                float mx  = fmaxf(logit, other);
                float e0  = expf(logit - mx);
                float e1  = expf(other - mx);
                out[row * 2 + lr] = e0 / (e0 + e1);
            }
        }
    }
}

extern "C" void kernel_launch(void* const* d_in, const int* in_sizes, int n_in,
                              void* d_out, int out_size, void* d_ws, size_t ws_size,
                              hipStream_t stream) {
    const float* x  = (const float*)d_in[0];
    const float* W0 = (const float*)d_in[1];  const float* b0 = (const float*)d_in[2];
    const float* W1 = (const float*)d_in[3];  const float* b1 = (const float*)d_in[4];
    const float* W2 = (const float*)d_in[5];  const float* b2 = (const float*)d_in[6];
    const float* W3 = (const float*)d_in[7];  const float* b3 = (const float*)d_in[8];
    const float* U0 = (const float*)d_in[9];  const float* c0 = (const float*)d_in[10];
    const float* U1 = (const float*)d_in[11]; const float* c1 = (const float*)d_in[12];
    const float* U2 = (const float*)d_in[13]; const float* c2 = (const float*)d_in[14];
    const float* U3 = (const float*)d_in[15]; const float* c3 = (const float*)d_in[16];
    const float* U4 = (const float*)d_in[17]; const float* c4 = (const float*)d_in[18];
    float* out  = (float*)d_out;
    float* pbuf = (float*)d_ws;              // 32768 tiles x 8 f32 = 1 MB

    size_t lds1 = (size_t)(H_PAD_N * 32 + 2 * H_PAD_N * H_PAD_K + 16 * H_PAD_K
                           + 8 * 16 * ACT_STRIDE) * sizeof(_Float16);   // ~99 KB
    pfn_stage1<<<512, 256, lds1, stream>>>(x, W0, b0, W1, b1, W2, b2, W3, b3, pbuf);

    size_t lds2 = (size_t)(H_PAD_N * 32 + 3 * H_PAD_N * H_PAD_K + 16 * H_PAD_K
                           + 8 * 16 * ACT_STRIDE) * sizeof(_Float16);   // ~127 KB
    pfn_stage2<<<32, 256, lds2, stream>>>(pbuf, U0, c0, U1, c1, U2, c2, U3, c3,
                                          U4, c4, out);
}
